// BridgeLayer_14602888806494
// MI455X (gfx1250) — matrix operand order repaired and checked by reading, not compiled, not run on hardware
//
#include <hip/hip_runtime.h>

typedef __bf16 bf16;
typedef __attribute__((ext_vector_type(16))) __bf16 v16bf;
typedef __attribute__((ext_vector_type(8)))  float  v8f;

#define DEV static __device__ __forceinline__

// ---------------------------------------------------------------------------
// CDNA5 async LDS copy (GLOBAL_LOAD_ASYNC_TO_LDS_B128, ASYNCcnt-tracked)
// Per-lane: LDS[lds_addr] = MEM[gaddr], 16 bytes. Wait with s_wait_asynccnt.
// ---------------------------------------------------------------------------
DEV void async_ld_b128(const void* lds, const void* g) {
  unsigned lds_addr = (unsigned)(unsigned long long)lds;  // low 32 = LDS offset
  unsigned long long gaddr = (unsigned long long)g;
  asm volatile("global_load_async_to_lds_b128 %0, %1, off"
               :: "v"(lds_addr), "v"(gaddr) : "memory");
}
DEV void wait_async0() {
  asm volatile("s_wait_asynccnt 0x0" ::: "memory");
}

// ---------------------------------------------------------------------------
// WMMA helpers (gfx1250 wave32 layouts per CDNA5 ISA 7.12.2)
// ---------------------------------------------------------------------------
DEV v8f wmma_bf16(v16bf a, v16bf b, v8f c) {
  return __builtin_amdgcn_wmma_f32_16x16x32_bf16(false, a, false, b, (short)0, c,
                                                 false, false);
}

union frag_cast { uint4 u[2]; v16bf v; bf16 h[16]; };

// A-matrix 16x32 bf16 fragment from a row-major tile (stride ld, elements).
// a[0..15] = K = kh*16+0..15 => 2x ds_load_b128.
DEV v16bf load_a_frag(const bf16* p, int ld, int row0, int k0) {
  int lane = threadIdx.x & 31;
  int m = lane & 15, kh = lane >> 4;
  const bf16* r = p + (row0 + m) * ld + k0 + kh * 16;
  frag_cast f;
  f.u[0] = *(const uint4*)r;
  f.u[1] = *(const uint4*)(r + 8);
  return f.v;
}

// B-matrix 32x16 fragment from a PRE-SWIZZLED tile:
// element (k,n) of the tile lives at  tile*768 + ((k>>4)*16 + n)*24 + (k&15)
// (16 payload + 8 pad elements per lane => conflict-free 2x ds_load_b128).
DEV v16bf load_b_frag_sw(const bf16* p, int tile) {
  int lane = threadIdx.x & 31;
  const bf16* r = p + tile * 768 + lane * 24;
  frag_cast f;
  f.u[0] = *(const uint4*)r;
  f.u[1] = *(const uint4*)(r + 8);
  return f.v;
}

// ---------------------------------------------------------------------------
// One-time weight conversion f32 -> bf16, global fragment-swizzled layout:
// dst[((n>>4)*(K>>5) + (k>>5))*512 + (((k>>4)&1)*16 + (n&15))*16 + (k&15)]
// Each thread handles 16 consecutive k for one n -> 32B contiguous store.
// ---------------------------------------------------------------------------
__global__ void __launch_bounds__(256) convert_w_kernel(
    const float* __restrict__ src, bf16* __restrict__ dst, int K, int N) {
  int g = blockIdx.x * 256 + threadIdx.x;
  int n = g % N;
  int kt = g / N;          // 16-k group
  int k0 = kt * 16;
  frag_cast f;
#pragma unroll
  for (int i = 0; i < 16; ++i) f.h[i] = (bf16)src[(long)(k0 + i) * N + n];
  long tile = (long)(n >> 4) * (K >> 5) + (k0 >> 5);
  int lane16 = ((k0 >> 4) & 1) * 16 + (n & 15);
  uint4* d = (uint4*)(dst + tile * 512 + lane16 * 16);
  d[0] = f.u[0];
  d[1] = f.u[1];
}

// ---------------------------------------------------------------------------
// RMSNorm: one block per row, f32 in -> bf16 out
// ---------------------------------------------------------------------------
__global__ void __launch_bounds__(256) rmsnorm_kernel(
    const float* __restrict__ x, const float* __restrict__ w,
    bf16* __restrict__ y, int D) {
  __shared__ float red[256];
  int row = blockIdx.x, tid = threadIdx.x;
  const float* xr = x + (long)row * D;
  float ss = 0.f;
  for (int i = tid; i < D; i += 256) { float v = xr[i]; ss += v * v; }
  red[tid] = ss;
  __syncthreads();
  for (int s = 128; s > 0; s >>= 1) {
    if (tid < s) red[tid] += red[tid + s];
    __syncthreads();
  }
  float scale = rsqrtf(red[0] / (float)D + 1e-6f);
  bf16* yr = y + (long)row * D;
  for (int i = tid; i < D; i += 256) yr[i] = (bf16)(xr[i] * scale * w[i]);
}

// ---------------------------------------------------------------------------
// bf16 GEMM: C[M,N] = A[M,K](bf16) * Bsw (bf16, fragment-swizzled global).
// 128x128 block, 8 waves, each wave 32x64 (8 WMMA accumulators).
// A and B tiles staged with async b128 copies to LDS.
// ---------------------------------------------------------------------------
__global__ void __launch_bounds__(256) gemm_bf16_kernel(
    const bf16* __restrict__ A, const bf16* __restrict__ Bsw,
    int M, int N, int K,
    const float* __restrict__ res, const float* __restrict__ res2,
    float* __restrict__ outf, bf16* __restrict__ outb) {
  __shared__ __align__(16) bf16 As[128 * 40];   // padded row stride
  __shared__ __align__(16) bf16 Bs[8 * 768];    // 8 fragment-major 32x16 tiles
  int tid = threadIdx.x;
  int m0 = blockIdx.y * 128, n0 = blockIdx.x * 128;
  int wave = tid >> 5;
  int wr = wave >> 1, wc = wave & 1;  // 4x2 wave grid

  v8f acc[2][4];
#pragma unroll
  for (int i = 0; i < 2; ++i)
#pragma unroll
    for (int j = 0; j < 4; ++j)
      acc[i][j] = (v8f){0.f, 0.f, 0.f, 0.f, 0.f, 0.f, 0.f, 0.f};

  int arow = tid >> 1, acol = (tid & 1) * 16;        // A: 128 rows x 32 cols
  int btile = tid >> 5, bl = tid & 31;               // B: 8 tiles x 32 lanes
  const bf16* bsrc_base = Bsw + ((long)((n0 >> 4) + btile) * (K >> 5)) * 512 + bl * 16;
  bf16* bdst = Bs + btile * 768 + bl * 24;

  for (int k0 = 0; k0 < K; k0 += 32) {
    // async-stage A tile 128x32 (2x b128/thread)
    const bf16* ag = A + (long)(m0 + arow) * K + k0 + acol;
    bf16* ad = As + arow * 40 + acol;
    async_ld_b128(ad, ag);
    async_ld_b128(ad + 8, ag + 8);
    // async-stage B tile (pre-swizzled in global: straight copy, 2x b128)
    const bf16* bg = bsrc_base + (long)(k0 >> 5) * 512;
    async_ld_b128(bdst, bg);
    async_ld_b128(bdst + 8, bg + 8);
    if (k0 + 32 < K) {  // prefetch next k-step (global_prefetch_b8)
      __builtin_prefetch((const void*)(ag + 32), 0, 1);
      __builtin_prefetch((const void*)(bg + 512), 0, 1);
    }
    wait_async0();
    __syncthreads();
    v16bf a0 = load_a_frag(As, 40, wr * 32, 0);
    v16bf a1 = load_a_frag(As, 40, wr * 32 + 16, 0);
#pragma unroll
    for (int j = 0; j < 4; ++j) {
      v16bf b = load_b_frag_sw(Bs, wc * 4 + j);
      acc[0][j] = wmma_bf16(a0, b, acc[0][j]);
      acc[1][j] = wmma_bf16(a1, b, acc[1][j]);
    }
    __syncthreads();
  }

  int lane = tid & 31;
  int cn = lane & 15, kh = lane >> 4;
#pragma unroll
  for (int mt = 0; mt < 2; ++mt)
#pragma unroll
    for (int j = 0; j < 4; ++j)
#pragma unroll
      for (int i = 0; i < 8; ++i) {
        int row = m0 + wr * 32 + mt * 16 + i + 8 * kh;
        int col = n0 + wc * 64 + j * 16 + cn;
        long idx = (long)row * N + col;
        float v = acc[mt][j][i];
        if (res)  v += res[idx];
        if (res2) v += res2[idx];
        if (outf) outf[idx] = v;
        if (outb) outb[idx] = (bf16)v;
      }
}

// ---------------------------------------------------------------------------
// Per-token slot attention over N=8 + sparse top-4 readout.
// ---------------------------------------------------------------------------
__global__ void __launch_bounds__(256) attn_s_sparse_kernel(
    const bf16* __restrict__ Qb, const bf16* __restrict__ Kb,
    const bf16* __restrict__ Vb, const float* __restrict__ gate,
    float* __restrict__ gsp, bf16* __restrict__ aS) {
  __shared__ float dsum[8][8];
  __shared__ float wsel[8];
  int t = blockIdx.x, tid = threadIdx.x;
  int h = tid >> 5, lane = tid & 31;
  const float scale_s = 0.0883883476483184f;  // 128^-0.5

  float q[4];
  const bf16* qp = Qb + (long)t * 1024 + h * 128 + lane * 4;
#pragma unroll
  for (int j = 0; j < 4; ++j) q[j] = (float)qp[j];

  float dots[8];
#pragma unroll
  for (int nn = 0; nn < 8; ++nn) {
    const bf16* kp = Kb + ((long)t * 8 + nn) * 1024 + h * 128 + lane * 4;
    float d = 0.f;
#pragma unroll
    for (int j = 0; j < 4; ++j) d += q[j] * (float)kp[j];
    d += __shfl_xor(d, 16, 32);
    d += __shfl_xor(d, 8, 32);
    d += __shfl_xor(d, 4, 32);
    d += __shfl_xor(d, 2, 32);
    d += __shfl_xor(d, 1, 32);
    dots[nn] = d;
  }
  if (lane == 0) {
#pragma unroll
    for (int nn = 0; nn < 8; ++nn) dsum[h][nn] = dots[nn];
  }
  float mx = dots[0];
#pragma unroll
  for (int nn = 1; nn < 8; ++nn) mx = fmaxf(mx, dots[nn]);
  float e[8], ssum = 0.f;
#pragma unroll
  for (int nn = 0; nn < 8; ++nn) {
    e[nn] = __expf((dots[nn] - mx) * scale_s);
    ssum += e[nn];
  }
  float inv = 1.f / ssum;
  float acc[4] = {0.f, 0.f, 0.f, 0.f};
#pragma unroll
  for (int nn = 0; nn < 8; ++nn) {
    float wn = e[nn] * inv;
    const bf16* vp = Vb + ((long)t * 8 + nn) * 1024 + h * 128 + lane * 4;
#pragma unroll
    for (int j = 0; j < 4; ++j) acc[j] += wn * (float)vp[j];
  }
  bf16* op = aS + (long)t * 1024 + h * 128 + lane * 4;
#pragma unroll
  for (int j = 0; j < 4; ++j) op[j] = (bf16)acc[j];
  __syncthreads();

  if (tid == 0) {
    float sp[8];
#pragma unroll
    for (int nn = 0; nn < 8; ++nn) {
      float s = 0.f;
#pragma unroll
      for (int hh = 0; hh < 8; ++hh) s += dsum[hh][nn];
      sp[nn] = s * scale_s;
    }
    bool used[8] = {false, false, false, false, false, false, false, false};
    float vals[4];
    int idx[4];
#pragma unroll
    for (int k = 0; k < 4; ++k) {
      float best = -1e30f;
      int bi = 0;
      for (int nn = 0; nn < 8; ++nn)
        if (!used[nn] && sp[nn] > best) { best = sp[nn]; bi = nn; }
      used[bi] = true;
      vals[k] = best;
      idx[k] = bi;
    }
    float m2 = vals[0], es[4], ss = 0.f;
#pragma unroll
    for (int k = 0; k < 4; ++k) { es[k] = __expf(vals[k] - m2); ss += es[k]; }
#pragma unroll
    for (int nn = 0; nn < 8; ++nn) wsel[nn] = 0.f;
#pragma unroll
    for (int k = 0; k < 4; ++k) wsel[idx[k]] = es[k] / ss;
  }
  __syncthreads();

  float g = gate[t];
  int d0 = tid * 4;  // 256 threads * 4 = 1024
  float a2[4] = {0.f, 0.f, 0.f, 0.f};
#pragma unroll
  for (int nn = 0; nn < 8; ++nn) {
    float wn = wsel[nn];
    const bf16* vp = Vb + ((long)t * 8 + nn) * 1024 + d0;
#pragma unroll
    for (int j = 0; j < 4; ++j) a2[j] += wn * (float)vp[j];
  }
#pragma unroll
  for (int j = 0; j < 4; ++j) gsp[(long)t * 1024 + d0 + j] = g * a2[j];
}

// ---------------------------------------------------------------------------
// Causal flash attention for the (n,h) heads: 64 heads, T=1024, d=64.
// Block = (q-block of 128 rows, n, h); 8 waves, 16 q-rows each.
// Q and K^T tiles staged with async b128 copies; V scattered fragment-major.
// ---------------------------------------------------------------------------
__global__ void __launch_bounds__(256) flash_attn_r_kernel(
    const bf16* __restrict__ QR, const bf16* __restrict__ KR,
    const bf16* __restrict__ VR, bf16* __restrict__ aR) {
  __shared__ __align__(16) bf16 qs[128 * 72];     // Q tile, padded stride
  __shared__ __align__(16) bf16 ks[4 * 768];      // K^T fragment tiles
  __shared__ __align__(16) bf16 vs[4 * 768];      // V fragment tiles
  __shared__ __align__(16) bf16 ps[8 * 16 * 40];  // per-wave P tile, padded
  int qb = blockIdx.x, n = blockIdx.y, h = blockIdx.z;
  int tid = threadIdx.x, wave = tid >> 5, lane = tid & 31;
  int qbase = qb * 128;
  const float scale_r = 0.125f;  // 64^-0.5

  {  // async-stage Q tile (128 x 64): 4x b128/thread
    int row = tid >> 1, colb = (tid & 1) * 32;
    const bf16* qg = QR + ((long)(qbase + row) * 8 + n) * 512 + h * 64 + colb;
    bf16* ql = qs + row * 72 + colb;
#pragma unroll
    for (int j = 0; j < 4; ++j) async_ld_b128(ql + j * 8, qg + j * 8);
  }

  v8f o[4];
#pragma unroll
  for (int j = 0; j < 4; ++j)
    o[j] = (v8f){0.f, 0.f, 0.f, 0.f, 0.f, 0.f, 0.f, 0.f};
  float rm[8], rl[8];
#pragma unroll
  for (int i = 0; i < 8; ++i) { rm[i] = -1e30f; rl[i] = 0.f; }

  int cn = lane & 15, kh = lane >> 4;
  // K/V staging decomposition: thread loads row kr, 8 d-columns at c..c+7
  int kr = tid >> 3, c = (tid & 7) * 8;
  int kkt = kr >> 4, kn = kr & 15;              // key tile / key within tile
  int kdstep = c >> 5, kkh = (c >> 4) & 1;      // d-chunk / d-half
  int vjn = c >> 4, vp = kr & 15, vkh = kr >> 4;
  bf16* kdst = ks + (kdstep * 2 + kkt) * 768 + (kkh * 16 + kn) * 24 + (c & 15);
  bf16* vdst = vs + vjn * 768 + (vkh * 16 + (c & 15)) * 24 + vp;
  int nkb = (qb + 1) * 4;  // causal: keys up to (qb+1)*128

  for (int kb = 0; kb < nkb; ++kb) {
    int kbase = kb * 32;
    __syncthreads();
    // K^T tile: fragment element p runs along d => straight async b128 copy
    async_ld_b128(kdst, KR + (long)(kbase + kr) * 512 + h * 64 + c);
    {  // V tile: fragment element p runs along keys => scatter 8 b16 stores
      uint4 vv = *(const uint4*)(VR + (long)(kbase + kr) * 512 + h * 64 + c);
      const bf16* ve = (const bf16*)&vv;
#pragma unroll
      for (int jj = 0; jj < 8; ++jj) vdst[jj * 24] = ve[jj];
    }
    wait_async0();
    __syncthreads();

    if (kbase <= qbase + wave * 16 + 15) {  // skip fully-masked wave rows
      v16bf aq0 = load_a_frag(qs, 72, wave * 16, 0);
      v16bf aq1 = load_a_frag(qs, 72, wave * 16, 32);
      v8f s0 = (v8f){0.f, 0.f, 0.f, 0.f, 0.f, 0.f, 0.f, 0.f};
      v8f s1 = s0;
      s0 = wmma_bf16(aq0, load_b_frag_sw(ks, 0), s0);
      s0 = wmma_bf16(aq1, load_b_frag_sw(ks, 2), s0);
      s1 = wmma_bf16(aq0, load_b_frag_sw(ks, 1), s1);
      s1 = wmma_bf16(aq1, load_b_frag_sw(ks, 3), s1);

#pragma unroll
      for (int i = 0; i < 8; ++i) {
        int tq = qbase + wave * 16 + i + 8 * kh;
        float x0 = (kbase + cn <= tq) ? s0[i] * scale_r : -1e30f;
        float x1 = (kbase + 16 + cn <= tq) ? s1[i] * scale_r : -1e30f;
        float mxv = fmaxf(x0, x1);
        mxv = fmaxf(mxv, __shfl_xor(mxv, 1, 32));
        mxv = fmaxf(mxv, __shfl_xor(mxv, 2, 32));
        mxv = fmaxf(mxv, __shfl_xor(mxv, 4, 32));
        mxv = fmaxf(mxv, __shfl_xor(mxv, 8, 32));
        float mn = fmaxf(rm[i], mxv);
        float al = __expf(rm[i] - mn);
        x0 = __expf(x0 - mn);
        x1 = __expf(x1 - mn);
        float rs = x0 + x1;
        rs += __shfl_xor(rs, 1, 32);
        rs += __shfl_xor(rs, 2, 32);
        rs += __shfl_xor(rs, 4, 32);
        rs += __shfl_xor(rs, 8, 32);
        rl[i] = rl[i] * al + rs;
        rm[i] = mn;
#pragma unroll
        for (int j = 0; j < 4; ++j) o[j][i] *= al;
        bf16* pw = ps + wave * 640 + (i + 8 * kh) * 40;
        pw[cn] = (bf16)x0;
        pw[16 + cn] = (bf16)x1;
      }
      v16bf pa = load_a_frag(ps + wave * 640, 40, 0, 0);
#pragma unroll
      for (int j = 0; j < 4; ++j)
        o[j] = wmma_bf16(pa, load_b_frag_sw(vs, j), o[j]);
    }
  }

#pragma unroll
  for (int i = 0; i < 8; ++i) {
    float inv = 1.f / rl[i];
    int tq = qbase + wave * 16 + i + 8 * kh;
#pragma unroll
    for (int j = 0; j < 4; ++j)
      aR[((long)tq * 8 + n) * 512 + h * 64 + j * 16 + cn] = (bf16)(o[j][i] * inv);
  }
}

// ---------------------------------------------------------------------------
// Host orchestration
// ---------------------------------------------------------------------------
extern "C" void kernel_launch(void* const* d_in, const int* in_sizes, int n_in,
                              void* d_out, int out_size, void* d_ws,
                              size_t ws_size, hipStream_t stream) {
  (void)in_sizes; (void)n_in; (void)out_size; (void)ws_size;
  const float* s    = (const float*)d_in[0];
  const float* r    = (const float*)d_in[1];
  const float* gate = (const float*)d_in[2];
  const float* nsw  = (const float*)d_in[3];
  const float* nrw  = (const float*)d_in[4];
  const float* Wsq  = (const float*)d_in[5];
  const float* Wsk  = (const float*)d_in[6];
  const float* Wsv  = (const float*)d_in[7];
  const float* Wso  = (const float*)d_in[8];
  const float* Wrq  = (const float*)d_in[9];
  const float* Wrk  = (const float*)d_in[10];
  const float* Wrv  = (const float*)d_in[11];
  const float* Wro  = (const float*)d_in[12];

  char* ws = (char*)d_ws;
  const size_t MB = 1ull << 20;
  bf16* s_n  = (bf16*)(ws + 0 * MB);     // 1024x1024  (2 MB)
  bf16* r_n  = (bf16*)(ws + 2 * MB);     // 8192x512   (8 MB)
  bf16* Qb   = (bf16*)(ws + 10 * MB);    // 1024x1024  (2 MB)
  bf16* Kb   = (bf16*)(ws + 12 * MB);    // 8192x1024  (16 MB)
  bf16* Vb   = (bf16*)(ws + 28 * MB);    // 8192x1024  (16 MB)
  bf16* QRb  = (bf16*)(ws + 44 * MB);    // 8192x512   (8 MB)
  bf16* KRb  = (bf16*)(ws + 52 * MB);    // 1024x512   (1 MB)
  bf16* VRb  = (bf16*)(ws + 53 * MB);    // 1024x512   (1 MB)
  bf16* aSb  = (bf16*)(ws + 54 * MB);    // 1024x1024  (2 MB)
  bf16* aRb  = (bf16*)(ws + 56 * MB);    // 8192x512   (8 MB)
  float* gsp = (float*)(ws + 64 * MB);   // 1024x1024  (4 MB)
  // pre-converted fragment-swizzled bf16 weights
  bf16* Wsq_b = (bf16*)(ws + 68 * MB);   // 2 MB
  bf16* Wsk_b = (bf16*)(ws + 70 * MB);   // 1 MB
  bf16* Wsv_b = (bf16*)(ws + 71 * MB);   // 1 MB
  bf16* Wso_b = (bf16*)(ws + 72 * MB);   // 2 MB
  bf16* Wrq_b = (bf16*)(ws + 74 * MB);   // 0.5 MB
  bf16* Wrk_b = (bf16*)(ws + 75 * MB);   // 1 MB
  bf16* Wrv_b = (bf16*)(ws + 76 * MB);   // 1 MB
  bf16* Wro_b = (bf16*)(ws + 77 * MB);   // 0.5 MB

  float* outS = (float*)d_out;            // s2 + gate*r_sparse  (1024x1024)
  float* outR = outS + 1024 * 1024;       // r2                  (8192x512)

  // 0. One-time weight conversion to fragment-swizzled bf16
  convert_w_kernel<<<256, 256, 0, stream>>>(Wsq, Wsq_b, 1024, 1024);
  convert_w_kernel<<<128, 256, 0, stream>>>(Wsk, Wsk_b, 512, 1024);
  convert_w_kernel<<<128, 256, 0, stream>>>(Wsv, Wsv_b, 512, 1024);
  convert_w_kernel<<<256, 256, 0, stream>>>(Wso, Wso_b, 1024, 1024);
  convert_w_kernel<<<64, 256, 0, stream>>>(Wrq, Wrq_b, 512, 512);
  convert_w_kernel<<<128, 256, 0, stream>>>(Wrk, Wrk_b, 1024, 512);
  convert_w_kernel<<<128, 256, 0, stream>>>(Wrv, Wrv_b, 1024, 512);
  convert_w_kernel<<<64, 256, 0, stream>>>(Wro, Wro_b, 512, 512);

  // 1. RMSNorm
  rmsnorm_kernel<<<1024, 256, 0, stream>>>(s, nsw, s_n, 1024);
  rmsnorm_kernel<<<8192, 256, 0, stream>>>(r, nrw, r_n, 512);

  // 2. Projections (bf16 WMMA)
  gemm_bf16_kernel<<<dim3(8, 8), 256, 0, stream>>>(s_n, Wsq_b, 1024, 1024, 1024,
                                                   nullptr, nullptr, nullptr, Qb);
  gemm_bf16_kernel<<<dim3(8, 64), 256, 0, stream>>>(r_n, Wsk_b, 8192, 1024, 512,
                                                    nullptr, nullptr, nullptr, Kb);
  gemm_bf16_kernel<<<dim3(8, 64), 256, 0, stream>>>(r_n, Wsv_b, 8192, 1024, 512,
                                                    nullptr, nullptr, nullptr, Vb);
  gemm_bf16_kernel<<<dim3(4, 64), 256, 0, stream>>>(r_n, Wrq_b, 8192, 512, 512,
                                                    nullptr, nullptr, nullptr, QRb);
  gemm_bf16_kernel<<<dim3(4, 8), 256, 0, stream>>>(s_n, Wrk_b, 1024, 512, 1024,
                                                   nullptr, nullptr, nullptr, KRb);
  gemm_bf16_kernel<<<dim3(4, 8), 256, 0, stream>>>(s_n, Wrv_b, 1024, 512, 1024,
                                                   nullptr, nullptr, nullptr, VRb);

  // 3. Slot attention + sparse top-k readout (writes aSb, gsp = gate*r_sparse)
  attn_s_sparse_kernel<<<1024, 256, 0, stream>>>(Qb, Kb, Vb, gate, gsp, aSb);

  // 4. Causal flash attention over (n,h) heads
  flash_attn_r_kernel<<<dim3(8, 8, 8), 256, 0, stream>>>(QRb, KRb, VRb, aRb);

  // 5. Output projections with fused residuals -> d_out
  gemm_bf16_kernel<<<dim3(8, 8), 256, 0, stream>>>(aSb, Wso_b, 1024, 1024, 1024,
                                                   s, gsp, outS, nullptr);
  gemm_bf16_kernel<<<dim3(4, 64), 256, 0, stream>>>(aRb, Wro_b, 8192, 512, 512,
                                                    r, nullptr, outR, nullptr);
}
